// Spatial_27066883899532
// MI455X (gfx1250) — matrix operand order repaired and checked by reading, not compiled, not run on hardware
//
#include <hip/hip_runtime.h>
#include <hip/hip_bf16.h>

typedef __attribute__((ext_vector_type(16))) _Float16 v16h;
typedef __attribute__((ext_vector_type(8)))  _Float16 v8h;
typedef __attribute__((ext_vector_type(8)))  float    v8f;

#define NB    8
#define NC    12
#define NF    2
#define NNODE 2048
#define DFEAT 24
#define DPAD  32
#define KSEL  16
#define SROW  2064   // 2048 + 16 pad floats per score row

// ---------------------------------------------------------------------------
// Phase 1: build L2-normalized feature rows packed as [b][n][32] f16 (zero-
// padded K 24->32), and emit tgt_out[b][c][n] = x_c[b,c,flow,n].
// ---------------------------------------------------------------------------
__global__ __launch_bounds__(256)
void phase1_norm_pack(const float* __restrict__ xc, const int* __restrict__ flowp,
                      _Float16* __restrict__ fn16, float* __restrict__ tgt_out)
{
    int t = blockIdx.x * blockDim.x + threadIdx.x;   // 0 .. 8*2048-1
    int b = t >> 11;
    int n = t & (NNODE - 1);
    int flow = *flowp;

    float vals[DFEAT];
    float ss = 0.0f;
    #pragma unroll
    for (int f = 0; f < NF; ++f) {
        #pragma unroll
        for (int c = 0; c < NC; ++c) {
            float v = xc[(((size_t)(b * NC + c) * NF) + f) * NNODE + n];
            vals[f * NC + c] = v;
            ss += v * v;
        }
    }
    float inv = 1.0f / (sqrtf(ss) + 1e-8f);

    _Float16 h[DPAD];
    #pragma unroll
    for (int d = 0; d < DFEAT; ++d) h[d] = (_Float16)(vals[d] * inv);
    #pragma unroll
    for (int d = DFEAT; d < DPAD; ++d) h[d] = (_Float16)0.0f;

    _Float16* row = fn16 + (size_t)(b * NNODE + n) * DPAD;   // 64B aligned
    #pragma unroll
    for (int q = 0; q < 4; ++q)
        ((v8h*)row)[q] = ((const v8h*)h)[q];                 // 4x b128 stores

    // tgt_out[b][c][n] = x[b, flow, n, c] = vals[flow*12 + c]
    #pragma unroll
    for (int c = 0; c < NC; ++c)
        tgt_out[(size_t)(b * NC + c) * NNODE + n] = vals[flow * NC + c];
}

// monotone float->uint key: preserves ordering under unsigned compare
__device__ __forceinline__ unsigned int fkey(float f) {
    unsigned int u = __float_as_uint(f);
    return (u & 0x80000000u) ? ~u : (u | 0x80000000u);
}

// ---------------------------------------------------------------------------
// Phase 2: one block per (batch, 16-row tile). 8 waves.
//   Step A: each wave computes 16 of the 128 16x16 score tiles via
//           v_wmma_f32_16x16x32_f16 (K=24 padded to 32 -> ONE wmma per tile),
//           scattering D into a 16 x 2048 LDS score slab.
//   Step B: per row (2 rows/wave), 16 iterations of wave-cooperative argmax
//           with 64-bit packed keys (ties -> lowest index), knocking out the
//           winner with -inf and immediately gathering the 12 sx_c values.
// ---------------------------------------------------------------------------
__global__ __launch_bounds__(256)
void phase2_wmma_topk(const float* __restrict__ xc, const int* __restrict__ flowp,
                      const _Float16* __restrict__ fn16, float* __restrict__ sx)
{
    extern __shared__ float scores[];                 // 16 * SROW floats
    const int b    = blockIdx.x >> 7;                 // batch 0..7
    const int rt   = blockIdx.x & 127;                // row tile 0..127
    const int n0   = rt * 16;
    const int wave = threadIdx.x >> 5;
    const int lane = threadIdx.x & 31;
    const int sel  = lane >> 4;                       // half-wave select
    const int lr   = lane & 15;
    const int flow = *flowp;

    const _Float16* fb = fn16 + (size_t)b * NNODE * DPAD;

    // ---- A fragment (shared by every column tile this wave computes) ----
    // lane<16: row n0+lr, K{0..7} @ +0,  K{8..15} @ +16B
    // lane>=16: row n0+lr, K{16..23} @ +32B, K{24..31} @ +48B
    const _Float16* arow = fb + (size_t)(n0 + lr) * DPAD + sel * 16;
    union { v16h v; v8h h[2]; } au;
    au.h[0] = *(const v8h*)(arow);
    au.h[1] = *(const v8h*)(arow + 8);
    const v16h afrag = au.v;

    // ---- Step A: score tiles ----
    for (int i = 0; i < 16; ++i) {
        const int ct = wave + (i << 3);               // column tile 0..127
        const int m0 = ct * 16;
        // B frag: lane<16 -> col m0+lr K0..15 ; lane>=16 -> K16..31
        const _Float16* brow = fb + (size_t)(m0 + lr) * DPAD + sel * 16;
        const v16h bfrag = *(const v16h*)brow;

        v8f c = {};
        c = __builtin_amdgcn_wmma_f32_16x16x32_f16(
                /*neg_a=*/false, afrag, /*neg_b=*/false, bfrag,
                /*c_mod=*/(short)0, c, /*reuse_a=*/false, /*reuse_b=*/false);

        #pragma unroll
        for (int v = 0; v < 8; ++v) {
            const int M = v + (sel << 3);             // local row 0..15
            scores[M * SROW + m0 + lr] = c[v];
        }
    }
    __syncthreads();

    // ---- Step B: top-16 per row (8 waves x 2 rows, uniform control flow) ----
    for (int rr = 0; rr < 2; ++rr) {
        const int row   = wave * 2 + rr;              // 0..15
        float*    srow  = scores + row * SROW;
        const int nglob = n0 + row;

        for (int k = 0; k < KSEL; ++k) {
            unsigned long long best = 0ull;
            for (int m = lane; m < NNODE; m += 32) {
                unsigned long long key =
                    ((unsigned long long)fkey(srow[m]) << 32) |
                    (unsigned int)(NNODE - 1 - m);    // ties -> smaller index
                if (key > best) best = key;
            }
            #pragma unroll
            for (int off = 16; off > 0; off >>= 1) {
                unsigned long long o = __shfl_xor(best, off, 32);
                if (o > best) best = o;
            }
            const int widx = (NNODE - 1) - (int)(best & 0xFFFFFFFFu);

            if (lane == 0) srow[widx] = -3.0e38f;     // knock out winner
            if (lane < NC) {                          // gather 12 channels
                sx[((size_t)(b * NNODE + nglob) * KSEL + k) * NC + lane] =
                    xc[(((size_t)(b * NC + lane) * NF) + flow) * NNODE + widx];
            }
            __syncthreads();                          // uniform across block
        }
    }
}

// ---------------------------------------------------------------------------
extern "C" void kernel_launch(void* const* d_in, const int* in_sizes, int n_in,
                              void* d_out, int out_size, void* d_ws, size_t ws_size,
                              hipStream_t stream)
{
    (void)in_sizes; (void)n_in; (void)out_size; (void)ws_size;

    const float* xc    = (const float*)d_in[0];
    const int*   flowp = (const int*)d_in[1];

    float* out = (float*)d_out;
    float* sx  = out;                                          // (8,2048,16,12)
    float* tgt = out + (size_t)NB * NNODE * KSEL * NC;         // (8,12,2048)

    _Float16* fn16 = (_Float16*)d_ws;                          // 1 MB packed feats

    // Phase 1: 8*2048 nodes, one thread each
    phase1_norm_pack<<<(NB * NNODE) / 256, 256, 0, stream>>>(xc, flowp, fn16, tgt);

    // Phase 2: 8 batches * 128 row tiles; 16 x SROW f32 score slab in LDS
    const size_t smem = (size_t)16 * SROW * sizeof(float);     // 132 KB (< 320 KB/WGP)
    phase2_wmma_topk<<<NB * (NNODE / 16), 256, smem, stream>>>(xc, flowp, fn16, sx);
}
